// PointNetpp_58110907514885
// MI455X (gfx1250) — matrix operand order repaired and checked by reading, not compiled, not run on hardware
//
#include <hip/hip_runtime.h>

// ---------------------------------------------------------------------------
// PointNet++ forward for MI455X (gfx1250, wave32, WMMA bf16 16x16x32).
// Weights are pre-packed into WMMA B-fragment layout (bf16, zero padded) so
// the inner loops are pure vector loads + v_wmma, no per-element predication.
// ---------------------------------------------------------------------------

typedef __attribute__((ext_vector_type(16))) __bf16 v16bf;
typedef __attribute__((ext_vector_type(8)))  float  v8f;

#define BB   16      // batch
#define N1   8192    // points
#define S1   512
#define K1g  32
#define S2   128
#define K2g  64

__device__ __forceinline__ unsigned short f2bf(float f) {
  unsigned u = __float_as_uint(f);
  u += 0x7FFFu + ((u >> 16) & 1u);          // round-to-nearest-even
  return (unsigned short)(u >> 16);
}
__device__ __forceinline__ float bf2f(unsigned short u) {
  return __uint_as_float((unsigned)u << 16);
}

// A-fragment: two 16B chunks 32B apart (K = koff..koff+7, koff+16..koff+23)
__device__ __forceinline__ v16bf frag_split(const unsigned short* p) {
  union { v16bf v; uint4 q[2]; } f;
  f.q[0] = *(const uint4*)(p);
  f.q[1] = *(const uint4*)(p + 16);
  return f.v;
}
// B-fragment: one contiguous 32B chunk per lane (packed layout)
__device__ __forceinline__ v16bf frag_contig(const unsigned short* p) {
  union { v16bf v; uint4 q[2]; } f;
  f.q[0] = *(const uint4*)(p);
  f.q[1] = *(const uint4*)(p + 8);
  return f.v;
}

// ---------------------------------------------------------------- utilities
__global__ void k_zero(float* __restrict__ p, int n) {
  int i = blockIdx.x * blockDim.x + threadIdx.x;
  if (i < n) p[i] = 0.0f;
}

// points[B][6][N] -> xyz[B][N][3], norms[B][N][3]
__global__ void k_split(const float* __restrict__ pts, float* __restrict__ xyz,
                        float* __restrict__ nrm) {
  int e = blockIdx.x * blockDim.x + threadIdx.x;
  if (e >= BB * 6 * N1) return;
  int b = e / (6 * N1);
  int r = e - b * 6 * N1;
  int ch = r / N1, n = r - ch * N1;
  float v = pts[e];
  if (ch < 3) xyz[(b * N1 + n) * 3 + ch] = v;
  else        nrm[(b * N1 + n) * 3 + (ch - 3)] = v;
}

// Pack W[C x Dreal] (f32, row-major) into B-fragment layout:
// Wp[((tn*nkc + kc)*32 + lane)*16 + i] = bf16(W[k*Dreal+col]) (0 if OOB)
//   col = tn*16 + (lane&15), k = kc*32 + ((lane&16)?8:0) + (i<8 ? i : i+8)
__global__ void k_packw(const float* __restrict__ W, unsigned short* __restrict__ Wp,
                        int C, int Dreal, int nkc, int tilesN) {
  int e = blockIdx.x * blockDim.x + threadIdx.x;
  int total = tilesN * nkc * 512;
  if (e >= total) return;
  int i = e & 15;
  int lane = (e >> 4) & 31;
  int frag = e >> 9;
  int kc = frag % nkc, tn = frag / nkc;
  int col = tn * 16 + (lane & 15);
  int k = kc * 32 + ((lane & 16) ? 8 : 0) + (i < 8 ? i : i + 8);
  float v = (k < C && col < Dreal) ? W[k * Dreal + col] : 0.0f;
  Wp[e] = f2bf(v);
}

// --------------------------------------------------------- farthest sampling
#define FPS_BD 1024
__global__ __launch_bounds__(FPS_BD)
void k_fps(const float* __restrict__ xyz, int N, int npoint, int* __restrict__ fidx) {
  int b = blockIdx.x, tid = threadIdx.x;
  __shared__ float sv[FPS_BD];
  __shared__ int   si[FPS_BD];
  __shared__ float c3[3];
  __shared__ int   sfar;
  const int P = (N + FPS_BD - 1) / FPS_BD;          // <= 8
  float lx[8], ly[8], lz[8], md[8];
  for (int p = 0; p < P; ++p) {
    int i = tid + p * FPS_BD;
    if (i < N) {
      const float* q = xyz + (b * N + i) * 3;
      lx[p] = q[0]; ly[p] = q[1]; lz[p] = q[2];
    }
    md[p] = 3.0e38f;
  }
  if (tid == 0) sfar = 0;
  __syncthreads();
  for (int it = 0; it < npoint; ++it) {
    int far = sfar;
    if (tid == 0) fidx[b * npoint + it] = far;      // emit previous 'far'
    if (tid < 3)  c3[tid] = xyz[(b * N + far) * 3 + tid];
    __syncthreads();
    float cx = c3[0], cy = c3[1], cz = c3[2];
    float bestv = -3.0e38f; int besti = 0;
    for (int p = 0; p < P; ++p) {
      int i = tid + p * FPS_BD;
      if (i < N) {
        float dx = lx[p] - cx, dy = ly[p] - cy, dz = lz[p] - cz;
        float d = dx * dx + dy * dy + dz * dz;
        float nm = fminf(md[p], d);
        md[p] = nm;
        if (nm > bestv || (nm == bestv && i < besti)) { bestv = nm; besti = i; }
      }
    }
    sv[tid] = bestv; si[tid] = besti;
    __syncthreads();
    for (int off = FPS_BD / 2; off > 0; off >>= 1) {
      if (tid < off) {
        float ov = sv[tid + off]; int oi = si[tid + off];
        if (ov > sv[tid] || (ov == sv[tid] && oi < si[tid])) { sv[tid] = ov; si[tid] = oi; }
      }
      __syncthreads();
    }
    if (tid == 0) sfar = si[0];
    __syncthreads();
  }
}

// gather centers: nxyz[b][s][c] = xyz[b][fidx[b][s]][c]
__global__ void k_gather3(const float* __restrict__ xyz, const int* __restrict__ fidx,
                          float* __restrict__ nxyz, int N, int S) {
  int e = blockIdx.x * blockDim.x + threadIdx.x;
  if (e >= BB * S * 3) return;
  int c = e % 3, bs = e / 3;
  nxyz[e] = xyz[((bs / S) * N + fidx[bs]) * 3 + c];
}

// ------------------------------------------------------------- ball query
__global__ __launch_bounds__(32)
void k_ball(const float* __restrict__ nxyz, const float* __restrict__ xyz,
            int* __restrict__ gidx, int N, int S, int K, float radius) {
  int bs = blockIdx.x;                 // b*S + s
  int b = bs / S;
  int lane = threadIdx.x;
  float cx = nxyz[bs * 3 + 0], cy = nxyz[bs * 3 + 1], cz = nxyz[bs * 3 + 2];
  float r2 = radius * radius;
  __shared__ int sFlags[32];
  __shared__ int sCnt, sFirst;
  if (lane == 0) { sCnt = 0; sFirst = -1; }
  __syncthreads();
  int* row = gidx + bs * K;
  for (int j0 = 0; j0 < N; j0 += 32) {
    int j = j0 + lane;
    int pred = 0;
    if (j < N) {
      const float* q = xyz + (b * N + j) * 3;
      float dx = q[0] - cx, dy = q[1] - cy, dz = q[2] - cz;
      pred = (dx * dx + dy * dy + dz * dz) <= r2;
    }
    sFlags[lane] = pred;
    __syncthreads();
    int pre = 0;
    for (int t = 0; t < lane; ++t) pre += sFlags[t];
    int cnt = sCnt;
    if (pred && cnt + pre < K) row[cnt + pre] = j;
    __syncthreads();
    if (lane == 0) {
      int tot = 0, f = -1;
      for (int t = 0; t < 32; ++t) if (sFlags[t]) { ++tot; if (f < 0) f = j0 + t; }
      if (sFirst < 0 && f >= 0) sFirst = f;
      sCnt += tot;
    }
    __syncthreads();
    if (sCnt >= K) break;
  }
  int start = sCnt < K ? sCnt : K;
  int first = sFirst < 0 ? 0 : sFirst;
  for (int slot = start + lane; slot < K; slot += 32) row[slot] = first;
}

// ------------------------------------------------ WMMA bf16 layer (from LDS)
// A 16x32 bf16: lane<16 -> row=lane, K={koff+i, koff+16+i}; koff=8 for lane>=16
// B: pre-packed fragments (contiguous per lane). C/D: col=lane&15, row=r+half*8
__device__ __forceinline__ void mlp_layer(
    const unsigned short* __restrict__ src, int cs,
    const unsigned short* __restrict__ Wp, const float* __restrict__ scl,
    const float* __restrict__ bia, int D, int Min,
    unsigned short* __restrict__ dst, float* __restrict__ outRow) {
  int lane = threadIdx.x & 31, wv = threadIdx.x >> 5, nw = blockDim.x >> 5;
  int tilesM = Min >> 4, tilesN = D >> 4, nt = tilesM * tilesN;
  int nkc = cs >> 5;
  int koff = (lane & 16) ? 8 : 0;
  int lan15 = lane & 15;
  for (int t = wv; t < nt; t += nw) {
    int tm = t % tilesM, tn = t / tilesM;
    int row = tm * 16 + lan15;
    int col = tn * 16 + lan15;
    const unsigned short* ap = src + row * cs + koff;
    const unsigned short* bp = Wp + ((size_t)(tn * nkc) << 9) + (lane << 4);
    v8f acc = {};
    for (int kc = 0; kc < nkc; ++kc) {
      v16bf a = frag_split(ap + (kc << 5));
      v16bf bm = frag_contig(bp + (kc << 9));
      acc = __builtin_amdgcn_wmma_f32_16x16x32_bf16(
          false, a, false, bm, (short)0, acc, false, false);
    }
    float sc = scl[col], bb = bia[col];
    if (dst) {
#pragma unroll
      for (int r = 0; r < 8; ++r) {
        int rr = tm * 16 + r + ((lane & 16) ? 8 : 0);
        float v = fmaxf(acc[r] * sc + bb, 0.0f);
        dst[rr * D + col] = f2bf(v);
      }
    } else {                                   // last layer: max over rows
      float m = 0.0f;
#pragma unroll
      for (int r = 0; r < 8; ++r) m = fmaxf(m, fmaxf(acc[r] * sc + bb, 0.0f));
      m = fmaxf(m, __shfl_xor(m, 16, 32));     // join lane halves (16 rows)
      if (lane < 16)                           // relu>=0: int max == float max
        atomicMax((int*)(outRow + col), __float_as_int(m));
    }
  }
}

// ------------------------------- fused set-abstraction: gather+MLPx3+max
__global__ __launch_bounds__(256)
void k_sa_fused(const float* __restrict__ xyz, const float* __restrict__ feats,
                const float* __restrict__ nxyz, const int* __restrict__ gidx,
                int N, int S, int Kg, int Cfeat, int Cpad,
                const unsigned short* w1, const float* sc1, const float* bi1, int d1,
                const unsigned short* w2, const float* sc2, const float* bi2, int d2,
                const unsigned short* w3, const float* sc3, const float* bi3, int d3,
                float* __restrict__ outF) {
  __shared__ unsigned short sIn[64 * 160];   // up to K=64, Cpad=160 (bf16)
  __shared__ unsigned short sA[64 * 128];
  __shared__ unsigned short sBv[64 * 128];
  int bs = blockIdx.x;
  int b = bs / S, s = bs % S;
  int Cin = 3 + Cfeat;
  for (int e = threadIdx.x; e < Kg * Cpad; e += blockDim.x) {
    int j = e / Cpad, c = e - j * Cpad;
    float v = 0.0f;
    if (c < Cin) {
      int idx = gidx[bs * Kg + j];
      if (c < 3) v = xyz[(b * N + idx) * 3 + c] - nxyz[bs * 3 + c];
      else       v = feats[(b * N + idx) * Cfeat + (c - 3)];
    }
    sIn[j * Cpad + c] = f2bf(v);
  }
  __syncthreads();
  mlp_layer(sIn, Cpad, w1, sc1, bi1, d1, Kg, sA, nullptr);
  __syncthreads();
  mlp_layer(sA, d1, w2, sc2, bi2, d2, Kg, sBv, nullptr);
  __syncthreads();
  mlp_layer(sBv, d2, w3, sc3, bi3, d3, Kg, nullptr, outF + bs * d3);
}

// --------------- generic WMMA GEMM, bf16 in / bf16 out (padded activations)
// Y = act((X@W + b0)*s + b1); X: M x Cs bf16 (Cs mult of 32, zero padded);
// Wp packed fragments covering D = tilesN*16 columns (zero padded >= Dreal).
__global__ __launch_bounds__(256)
void k_gemm_bf(const unsigned short* __restrict__ X, int Cs,
               const unsigned short* __restrict__ Wp,
               const float* __restrict__ b0, const float* __restrict__ scl,
               const float* __restrict__ b1,
               unsigned short* __restrict__ Ybf, float* __restrict__ Yf32,
               int M, int D, int Dreal, int relu) {
  int lane = threadIdx.x & 31, wv = threadIdx.x >> 5, nw = blockDim.x >> 5;
  int tilesM = M >> 4, tilesN = D >> 4;
  int nt = tilesM * tilesN;
  int gt = blockIdx.x * nw + wv;
  if (gt >= nt) return;
  int tm = gt % tilesM, tn = gt / tilesM;
  int lan15 = lane & 15;
  int row = tm * 16 + lan15;
  int col = tn * 16 + lan15;
  int koff = (lane & 16) ? 8 : 0;
  int nkc = Cs >> 5;
  const unsigned short* ap = X + row * Cs + koff;
  const unsigned short* bp = Wp + ((size_t)(tn * nkc) << 9) + (lane << 4);
  v8f acc = {};
  for (int kc = 0; kc < nkc; ++kc) {
    v16bf a = frag_split(ap + (kc << 5));
    v16bf bm = frag_contig(bp + (kc << 9));
    acc = __builtin_amdgcn_wmma_f32_16x16x32_bf16(
        false, a, false, bm, (short)0, acc, false, false);
  }
  if (col >= Dreal) return;
  float ba = b0 ? b0[col] : 0.0f;
  float sc = scl ? scl[col] : 1.0f;
  float bb = b1 ? b1[col] : 0.0f;
#pragma unroll
  for (int r = 0; r < 8; ++r) {
    int rr = tm * 16 + r + ((lane & 16) ? 8 : 0);
    float v = (acc[r] + ba) * sc + bb;
    if (relu) v = fmaxf(v, 0.0f);
    if (Ybf)  Ybf[rr * D + col] = f2bf(v);
    if (Yf32) Yf32[rr * Dreal + col] = v;
  }
}

// x3 (bf16, 2048 x 288 zero-padded) = concat(nxyz2, f2)
__global__ void k_buildx3(const float* __restrict__ nxyz2, const float* __restrict__ f2,
                          unsigned short* __restrict__ x3) {
  int e = blockIdx.x * blockDim.x + threadIdx.x;
  if (e >= BB * S2 * 288) return;
  int row = e / 288, c = e - row * 288;
  float v = 0.0f;
  if (c < 3)        v = nxyz2[row * 3 + c];
  else if (c < 259) v = f2[row * 256 + (c - 3)];
  x3[e] = f2bf(v);
}

// g[b][d] = max_j y[(b*128+j)][d]   (bf16 in, bf16 out, d < 1024)
__global__ void k_maxrows(const unsigned short* __restrict__ y,
                          unsigned short* __restrict__ g) {
  int e = blockIdx.x * blockDim.x + threadIdx.x;
  if (e >= BB * 1024) return;
  int b = e / 1024, d = e - b * 1024;
  float m = bf2f(y[(b * S2) * 1024 + d]);
  for (int j = 1; j < S2; ++j) m = fmaxf(m, bf2f(y[(b * S2 + j) * 1024 + d]));
  g[e] = f2bf(m);
}

// ---------------------------------------------------------------------------
extern "C" void kernel_launch(void* const* d_in, const int* in_sizes, int n_in,
                              void* d_out, int out_size, void* d_ws, size_t ws_size,
                              hipStream_t stream) {
  (void)in_sizes; (void)n_in; (void)out_size; (void)ws_size;
  // ---- inputs, flattened in setup_inputs() insertion order ----
  const float* points = (const float*)d_in[0];
  const float *s1w[3], *s1s[3], *s1b[3], *s2w[3], *s2s[3], *s2b[3], *s3w[3], *s3s[3], *s3b[3];
  for (int i = 0; i < 3; ++i) {
    s1w[i] = (const float*)d_in[1 + i];   s1s[i] = (const float*)d_in[4 + i];   s1b[i] = (const float*)d_in[7 + i];
    s2w[i] = (const float*)d_in[10 + i];  s2s[i] = (const float*)d_in[13 + i];  s2b[i] = (const float*)d_in[16 + i];
    s3w[i] = (const float*)d_in[19 + i];  s3s[i] = (const float*)d_in[22 + i];  s3b[i] = (const float*)d_in[25 + i];
  }
  const float* fc1_w = (const float*)d_in[28]; const float* fc1_b = (const float*)d_in[29];
  const float* bn1_s = (const float*)d_in[30]; const float* bn1_b = (const float*)d_in[31];
  const float* fc2_w = (const float*)d_in[32]; const float* fc2_b = (const float*)d_in[33];
  const float* bn2_s = (const float*)d_in[34]; const float* bn2_b = (const float*)d_in[35];
  const float* fc3_w = (const float*)d_in[36]; const float* fc3_b = (const float*)d_in[37];
  float* out = (float*)d_out;

  // ---- workspace carving ----
  char* base = (char*)d_ws;
  size_t cur = 0;
  auto carveF = [&](size_t nf) { float* p = (float*)(base + cur); cur = (cur + nf * 4 + 255) & ~(size_t)255; return p; };
  auto carveI = [&](size_t ni) { int*   p = (int*)(base + cur);   cur = (cur + ni * 4 + 255) & ~(size_t)255; return p; };
  auto carveH = [&](size_t nh) { unsigned short* p = (unsigned short*)(base + cur); cur = (cur + nh * 2 + 255) & ~(size_t)255; return p; };
  float* xyz   = carveF((size_t)BB * N1 * 3);
  float* nrm   = carveF((size_t)BB * N1 * 3);
  float* nxyz1 = carveF((size_t)BB * S1 * 3);
  float* nxyz2 = carveF((size_t)BB * S2 * 3);
  float* f1    = carveF((size_t)BB * S1 * 128);
  float* f2    = carveF((size_t)BB * S2 * 256);
  unsigned short* x3  = carveH((size_t)BB * S2 * 288);
  unsigned short* y3a = carveH((size_t)BB * S2 * 256);
  unsigned short* y3b = carveH((size_t)BB * S2 * 512);
  unsigned short* y3c = carveH((size_t)BB * S2 * 1024);
  unsigned short* gvec = carveH((size_t)BB * 1024);
  unsigned short* h1   = carveH((size_t)BB * 512);
  unsigned short* h2   = carveH((size_t)BB * 256);
  int* fidx1 = carveI((size_t)BB * S1);
  int* gidx1 = carveI((size_t)BB * S1 * K1g);
  int* fidx2 = carveI((size_t)BB * S2);
  int* gidx2 = carveI((size_t)BB * S2 * K2g);

  // packed weights: {Kpad, Dreal} per layer (fragment layout, bf16)
  struct PW { const float* W; int C, Dreal, Kpad; unsigned short* P; };
  PW pw[12] = {
    {s1w[0],    6,   64,   32, nullptr}, {s1w[1],   64,   64,   64, nullptr},
    {s1w[2],   64,  128,   64, nullptr}, {s2w[0],  131,  128,  160, nullptr},
    {s2w[1],  128,  128,  128, nullptr}, {s2w[2],  128,  256,  128, nullptr},
    {s3w[0],  259,  256,  288, nullptr}, {s3w[1],  256,  512,  256, nullptr},
    {s3w[2],  512, 1024,  512, nullptr}, {fc1_w,  1024,  512, 1024, nullptr},
    {fc2_w,   512,  256,  512, nullptr}, {fc3_w,   256,   10,  256, nullptr},
  };
  for (int i = 0; i < 12; ++i) {
    int tilesN = (pw[i].Dreal + 15) / 16;
    pw[i].P = carveH((size_t)pw[i].Kpad * tilesN * 16);
    int total = tilesN * (pw[i].Kpad / 32) * 512;
    k_packw<<<(total + 255) / 256, 256, 0, stream>>>(
        pw[i].W, pw[i].P, pw[i].C, pw[i].Dreal, pw[i].Kpad / 32, tilesN);
  }

  // ---- stage 0: split + zero the atomic-max outputs ----
  k_split<<<(BB * 6 * N1 + 255) / 256, 256, 0, stream>>>(points, xyz, nrm);
  k_zero<<<(BB * S1 * 128 + 255) / 256, 256, 0, stream>>>(f1, BB * S1 * 128);
  k_zero<<<(BB * S2 * 256 + 255) / 256, 256, 0, stream>>>(f2, BB * S2 * 256);

  // ---- SA1: fps(512) -> ball(r=0.2,K=32) -> fused MLP 6->64->64->128 + max
  k_fps<<<BB, FPS_BD, 0, stream>>>(xyz, N1, S1, fidx1);
  k_gather3<<<(BB * S1 * 3 + 255) / 256, 256, 0, stream>>>(xyz, fidx1, nxyz1, N1, S1);
  k_ball<<<BB * S1, 32, 0, stream>>>(nxyz1, xyz, gidx1, N1, S1, K1g, 0.2f);
  k_sa_fused<<<BB * S1, 256, 0, stream>>>(
      xyz, nrm, nxyz1, gidx1, N1, S1, K1g, /*Cfeat=*/3, /*Cpad=*/32,
      pw[0].P, s1s[0], s1b[0], 64,
      pw[1].P, s1s[1], s1b[1], 64,
      pw[2].P, s1s[2], s1b[2], 128, f1);

  // ---- SA2: fps(128) -> ball(r=0.4,K=64) -> 131->128->128->256 + max
  k_fps<<<BB, FPS_BD, 0, stream>>>(nxyz1, S1, S2, fidx2);
  k_gather3<<<(BB * S2 * 3 + 255) / 256, 256, 0, stream>>>(nxyz1, fidx2, nxyz2, S1, S2);
  k_ball<<<BB * S2, 32, 0, stream>>>(nxyz2, nxyz1, gidx2, S1, S2, K2g, 0.4f);
  k_sa_fused<<<BB * S2, 256, 0, stream>>>(
      nxyz1, f1, nxyz2, gidx2, S1, S2, K2g, /*Cfeat=*/128, /*Cpad=*/160,
      pw[3].P, s2s[0], s2b[0], 128,
      pw[4].P, s2s[1], s2b[1], 128,
      pw[5].P, s2s[2], s2b[2], 256, f2);

  // ---- SA3 (group-all): concat -> 259->256->512->1024 -> max over 128 ----
  k_buildx3<<<(BB * S2 * 288 + 255) / 256, 256, 0, stream>>>(nxyz2, f2, x3);
  int M3 = BB * S2; // 2048
  auto gemm = [&](const unsigned short* X, int Cs, const PW& w,
                  const float* b0, const float* sc, const float* b1,
                  unsigned short* Ybf, float* Yf32, int M, int relu) {
    int D = ((w.Dreal + 15) / 16) * 16;
    int tiles = (M / 16) * (D / 16);
    k_gemm_bf<<<(tiles + 7) / 8, 256, 0, stream>>>(
        X, Cs, w.P, b0, sc, b1, Ybf, Yf32, M, D, w.Dreal, relu);
  };
  gemm(x3,  288, pw[6], nullptr, s3s[0], s3b[0], y3a, nullptr, M3, 1);
  gemm(y3a, 256, pw[7], nullptr, s3s[1], s3b[1], y3b, nullptr, M3, 1);
  gemm(y3b, 512, pw[8], nullptr, s3s[2], s3b[2], y3c, nullptr, M3, 1);
  k_maxrows<<<(BB * 1024 + 255) / 256, 256, 0, stream>>>(y3c, gvec);

  // ---- FC head ----
  gemm(gvec, 1024, pw[9],  fc1_b, bn1_s, bn1_b, h1, nullptr, BB, 1);
  gemm(h1,    512, pw[10], fc2_b, bn2_s, bn2_b, h2, nullptr, BB, 1);
  gemm(h2,    256, pw[11], fc3_b, nullptr, nullptr, nullptr, out, BB, 0);
}
